// ContextQueryAttention_39230231281774
// MI455X (gfx1250) — matrix operand order repaired and checked by reading, not compiled, not run on hardware
//
#include <hip/hip_runtime.h>
#include <hip/hip_bf16.h>

typedef __attribute__((ext_vector_type(16))) _Float16 v16h;
typedef __attribute__((ext_vector_type(8)))  float    v8f;
typedef unsigned int u32x4 __attribute__((ext_vector_type(4)));
typedef int          i32x8 __attribute__((ext_vector_type(8)));
typedef int          i32x4 __attribute__((ext_vector_type(4)));

#define BB 32
#define HH 128
#define CC 2048
#define QQ 256
#define NEGV (-1e30f)

// ---------------- TDM: 2D tensor tile -> LDS (D# packing per CDNA5 ISA) ----
#if __has_builtin(__builtin_amdgcn_tensor_load_to_lds)
__device__ __forceinline__ void tdm_load_2d_f32(unsigned lds_off, const void* gptr,
                                                unsigned tensor_d0, unsigned tensor_d1,
                                                unsigned stride0_elems,
                                                unsigned tile_d0, unsigned tile_d1) {
  unsigned long long ga = (unsigned long long)(uintptr_t)gptr;
  u32x4 g0;
  g0[0] = 1u;                                             // count=1, user descriptor
  g0[1] = lds_off;                                        // lds_addr (bytes)
  g0[2] = (unsigned)(ga & 0xffffffffu);                   // global_addr[31:0]
  g0[3] = (unsigned)((ga >> 32) & 0x01ffffffu) | (2u << 30); // global_addr[56:32] | type=2
  i32x8 g1;
  g1[0] = (int)(2u << 16);                                // data_size=2 -> 4 bytes
  g1[1] = (int)((tensor_d0 & 0xffffu) << 16);             // tensor_dim0[15:0]
  g1[2] = (int)((tensor_d0 >> 16) | ((tensor_d1 & 0xffffu) << 16)); // td0 hi | td1 lo
  g1[3] = (int)((tensor_d1 >> 16) | (tile_d0 << 16));     // td1 hi | tile_dim0
  g1[4] = (int)(tile_d1 & 0xffffu);                       // tile_dim1, tile_dim2=0
  g1[5] = (int)stride0_elems;                             // tensor_dim0_stride[31:0]
  g1[6] = 0;                                              // stride0 hi | dim1_stride lo
  g1[7] = 0;
  i32x4 gz4 = {0, 0, 0, 0};
  i32x8 gz8 = {0, 0, 0, 0, 0, 0, 0, 0};
  __builtin_amdgcn_tensor_load_to_lds(g0, g1, gz4, gz4, gz8, 0);
}
#endif

// ---------------- K1: part_c[b,i] = ctx[b,:,i].w_c ; part_q[b,j] = qn[b,:,j].w_q ----------------
__global__ __launch_bounds__(256) void k_parts(const float* __restrict__ ctx,
                                               const float* __restrict__ qn,
                                               const float* __restrict__ w_c,
                                               const float* __restrict__ w_q,
                                               float* __restrict__ part_c,
                                               float* __restrict__ part_q) {
  int id = blockIdx.x * blockDim.x + threadIdx.x;
  if (id < BB * CC) {
    int b = id / CC, i = id % CC;
    const float* p = ctx + (size_t)b * HH * CC + i;
    float s = 0.f;
#pragma unroll 8
    for (int h = 0; h < HH; ++h) s += p[(size_t)h * CC] * w_c[h];
    part_c[id] = s;
  } else if (id < BB * CC + BB * QQ) {
    int id2 = id - BB * CC;
    int b = id2 / QQ, j = id2 % QQ;
    const float* p = qn + (size_t)b * HH * QQ + j;
    float s = 0.f;
#pragma unroll 8
    for (int h = 0; h < HH; ++h) s += p[(size_t)h * QQ] * w_q[h];
    part_q[id2] = s;
  }
}

// ---------------- K2: S = (ctx^T * w_cq) @ qn + part_c + part_q + bias ----------------
// Block (8 waves) computes a 64x64 tile. Tiles staged via TDM tensor_load_to_lds.
__global__ __launch_bounds__(256) void k_S(const float* __restrict__ ctx,
                                           const float* __restrict__ qn,
                                           const float* __restrict__ w_cq,
                                           const float* __restrict__ part_c,
                                           const float* __restrict__ part_q,
                                           const float* __restrict__ bias,
                                           float* __restrict__ S) {
  __shared__ float lds_c[HH * 64];   // [h][i] tile, 32 KB
  __shared__ float lds_q[HH * 64];   // [h][j] tile, 32 KB

  int blk = blockIdx.x;
  int b  = blk >> 7;                 // 128 tiles per batch: (C/64)*(Q/64)=32*4
  int t2 = blk & 127;
  int i0 = (t2 >> 2) << 6;
  int j0 = (t2 & 3) << 6;

#if __has_builtin(__builtin_amdgcn_tensor_load_to_lds)
  if (threadIdx.x < 32) {
    tdm_load_2d_f32((unsigned)(uintptr_t)(void*)lds_c,
                    ctx + (size_t)b * HH * CC + i0, CC, HH, CC, 64, HH);
    tdm_load_2d_f32((unsigned)(uintptr_t)(void*)lds_q,
                    qn + (size_t)b * HH * QQ + j0, QQ, HH, QQ, 64, HH);
    __builtin_amdgcn_s_wait_tensorcnt(0);
  }
  __syncthreads();
#else
  for (int e = threadIdx.x; e < HH * 64; e += 256) {
    int h = e >> 6, x = e & 63;
    lds_c[e] = ctx[(size_t)b * HH * CC + (size_t)h * CC + i0 + x];
    lds_q[e] = qn[(size_t)b * HH * QQ + (size_t)h * QQ + j0 + x];
  }
  __syncthreads();
#endif

  // pre-scale context tile rows by w_cq[h]
  for (int e = threadIdx.x; e < HH * 64; e += 256) lds_c[e] *= w_cq[e >> 6];
  __syncthreads();

  int wv = threadIdx.x >> 5;
  int lane = threadIdx.x & 31;
  int half = lane >> 4, r = lane & 15;
  float bv = bias[0];

  for (int s = wv; s < 16; s += 8) {         // 2 subtiles per wave, uniform per wave
    int isub = (s >> 2) << 4;
    int jsub = (s & 3) << 4;
    v8f acc = {};
#pragma unroll
    for (int k0 = 0; k0 < HH; k0 += 32) {
      v16h a, bf;
#pragma unroll
      for (int t = 0; t < 16; ++t) {
        int k  = k0 + t + (t & 8) + half * 8;       // A-fragment k
        a[t]   = (_Float16)lds_c[k * 64 + isub + r];
        int kb = k0 + t + (t & 8) + half * 8;       // B-fragment k
        bf[t]  = (_Float16)lds_q[kb * 64 + jsub + r];
      }
      acc = __builtin_amdgcn_wmma_f32_16x16x32_f16(false, a, false, bf, (short)0, acc, false, false);
    }
#pragma unroll
    for (int r8 = 0; r8 < 8; ++r8) {
      int i = i0 + isub + half * 8 + r8;
      int j = j0 + jsub + r;
      S[((size_t)b * CC + i) * QQ + j] = acc[r8] + part_c[b * CC + i] + part_q[b * QQ + j] + bv;
    }
  }
}

// ---------------- K3a: row softmax over Q (masked by question mask) -> f16 ----------------
__global__ __launch_bounds__(256) void k_row_softmax(const float* __restrict__ S,
                                                     const int* __restrict__ qmask,
                                                     _Float16* __restrict__ Srow) {
  __shared__ float red[256];
  int bi = blockIdx.x;            // b*C + i
  int b  = bi >> 11;              // / C
  int j  = threadIdx.x;
  float x = S[(size_t)bi * QQ + j];
  x = qmask[b * QQ + j] ? x : NEGV;
  red[j] = x; __syncthreads();
  for (int s = 128; s > 0; s >>= 1) { if (j < s) red[j] = fmaxf(red[j], red[j + s]); __syncthreads(); }
  float m = red[0]; __syncthreads();
  float e = __expf(x - m);
  red[j] = e; __syncthreads();
  for (int s = 128; s > 0; s >>= 1) { if (j < s) red[j] += red[j + s]; __syncthreads(); }
  float inv = 1.0f / red[0];
  Srow[(size_t)bi * QQ + j] = (_Float16)(e * inv);
}

// ---------------- K3b: column softmax over C (masked by context mask) -> transposed f16 ----------------
__global__ __launch_bounds__(256) void k_col_softmax(const float* __restrict__ S,
                                                     const int* __restrict__ cmask,
                                                     _Float16* __restrict__ ScolT) {
  __shared__ float red[256];
  int bj = blockIdx.x;            // b*Q + j
  int b  = bj >> 8;
  int j  = bj & 255;
  const float* Sb = S + (size_t)b * CC * QQ;
  const int* cm = cmask + b * CC;
  float x[8];
  float mx = NEGV;
#pragma unroll
  for (int c8 = 0; c8 < 8; ++c8) {
    int i = c8 * 256 + threadIdx.x;
    float v = Sb[(size_t)i * QQ + j];
    v = cm[i] ? v : NEGV;
    x[c8] = v; mx = fmaxf(mx, v);
  }
  red[threadIdx.x] = mx; __syncthreads();
  for (int s = 128; s > 0; s >>= 1) { if (threadIdx.x < s) red[threadIdx.x] = fmaxf(red[threadIdx.x], red[threadIdx.x + s]); __syncthreads(); }
  float m = red[0]; __syncthreads();
  float sum = 0.f;
#pragma unroll
  for (int c8 = 0; c8 < 8; ++c8) { x[c8] = __expf(x[c8] - m); sum += x[c8]; }
  red[threadIdx.x] = sum; __syncthreads();
  for (int s = 128; s > 0; s >>= 1) { if (threadIdx.x < s) red[threadIdx.x] += red[threadIdx.x + s]; __syncthreads(); }
  float inv = 1.0f / red[0];
  _Float16* outp = ScolT + ((size_t)b * QQ + j) * CC;
#pragma unroll
  for (int c8 = 0; c8 < 8; ++c8) { int i = c8 * 256 + threadIdx.x; outp[i] = (_Float16)(x[c8] * inv); }
}

// ---------------- K4: c2q[b,i,h] = sum_j Srow[b,i,j] * qn[b,h,j]; writes out0/out1/out2 ----------------
__global__ __launch_bounds__(128) void k_c2q(const _Float16* __restrict__ Srow,
                                             const float* __restrict__ qn,
                                             const float* __restrict__ ctx,
                                             float* __restrict__ out0,
                                             float* __restrict__ out1,
                                             float* __restrict__ out2) {
  int w    = blockIdx.x * 4 + (threadIdx.x >> 5);
  int lane = threadIdx.x & 31;
  int b  = w >> 10;               // 1024 tiles per batch: (C/16)*(H/16)=128*8
  int t2 = w & 1023;
  int i0 = (t2 >> 3) << 4;
  int h0 = (t2 & 7) << 4;
  int half = lane >> 4, r = lane & 15;
  const _Float16* Sb = Srow + (size_t)b * CC * QQ;
  const float* qnb   = qn   + (size_t)b * HH * QQ;
  v8f acc = {};
#pragma unroll
  for (int k0 = 0; k0 < QQ; k0 += 32) {
    v16h a, bf;
#pragma unroll
    for (int t = 0; t < 16; ++t) {
      int k  = k0 + t + (t & 8) + half * 8;
      a[t]   = Sb[(size_t)(i0 + r) * QQ + k];
      int kb = k0 + t + (t & 8) + half * 8;
      bf[t]  = (_Float16)(qnb[(size_t)(h0 + r) * QQ + kb]);
    }
    acc = __builtin_amdgcn_wmma_f32_16x16x32_f16(false, a, false, bf, (short)0, acc, false, false);
  }
#pragma unroll
  for (int r8 = 0; r8 < 8; ++r8) {
    int i = i0 + half * 8 + r8;
    int h = h0 + r;
    size_t idx = ((size_t)b * HH + h) * CC + i;
    float cv = ctx[idx];
    out0[idx] = cv;
    out1[idx] = acc[r8];
    out2[idx] = cv * acc[r8];
  }
}

// ---------------- K5: tmp[b,j,h] = sum_i ScolT[b,j,i] * ctx[b,h,i] ----------------
__global__ __launch_bounds__(128) void k_tmp(const _Float16* __restrict__ ScolT,
                                             const float* __restrict__ ctx,
                                             float* __restrict__ tmp) {
  int w    = blockIdx.x * 4 + (threadIdx.x >> 5);
  int lane = threadIdx.x & 31;
  int b  = w >> 7;                // 128 tiles per batch: (Q/16)*(H/16)=16*8
  int t2 = w & 127;
  int j0 = (t2 >> 3) << 4;
  int h0 = (t2 & 7) << 4;
  int half = lane >> 4, r = lane & 15;
  const _Float16* Sb = ScolT + (size_t)b * QQ * CC;
  const float* ctxb  = ctx   + (size_t)b * HH * CC;
  v8f acc = {};
  for (int k0 = 0; k0 < CC; k0 += 32) {
    v16h a, bf;
#pragma unroll
    for (int t = 0; t < 16; ++t) {
      int k  = k0 + t + (t & 8) + half * 8;
      a[t]   = Sb[(size_t)(j0 + r) * CC + k];
      int kb = k0 + t + (t & 8) + half * 8;
      bf[t]  = (_Float16)(ctxb[(size_t)(h0 + r) * CC + kb]);
    }
    acc = __builtin_amdgcn_wmma_f32_16x16x32_f16(false, a, false, bf, (short)0, acc, false, false);
  }
#pragma unroll
  for (int r8 = 0; r8 < 8; ++r8) {
    int j = j0 + half * 8 + r8;
    int h = h0 + r;
    tmp[((size_t)b * QQ + j) * HH + h] = acc[r8];
  }
}

// ---------------- K6: q2c[b,i,h] = sum_j Srow[b,i,j] * tmp[b,j,h]; out3 = ctx * q2c ----------------
__global__ __launch_bounds__(128) void k_q2c(const _Float16* __restrict__ Srow,
                                             const float* __restrict__ tmp,
                                             const float* __restrict__ ctx,
                                             float* __restrict__ out3) {
  int w    = blockIdx.x * 4 + (threadIdx.x >> 5);
  int lane = threadIdx.x & 31;
  int b  = w >> 10;
  int t2 = w & 1023;
  int i0 = (t2 >> 3) << 4;
  int h0 = (t2 & 7) << 4;
  int half = lane >> 4, r = lane & 15;
  const _Float16* Sb = Srow + (size_t)b * CC * QQ;
  const float* tb    = tmp  + (size_t)b * QQ * HH;
  v8f acc = {};
#pragma unroll
  for (int k0 = 0; k0 < QQ; k0 += 32) {
    v16h a, bf;
#pragma unroll
    for (int t = 0; t < 16; ++t) {
      int k  = k0 + t + (t & 8) + half * 8;
      a[t]   = Sb[(size_t)(i0 + r) * QQ + k];
      int kb = k0 + t + (t & 8) + half * 8;
      bf[t]  = (_Float16)(tb[(size_t)kb * HH + h0 + r]);
    }
    acc = __builtin_amdgcn_wmma_f32_16x16x32_f16(false, a, false, bf, (short)0, acc, false, false);
  }
#pragma unroll
  for (int r8 = 0; r8 < 8; ++r8) {
    int i = i0 + half * 8 + r8;
    int h = h0 + r;
    size_t idx = ((size_t)b * HH + h) * CC + i;
    out3[idx] = ctx[idx] * acc[r8];
  }
}

extern "C" void kernel_launch(void* const* d_in, const int* in_sizes, int n_in,
                              void* d_out, int out_size, void* d_ws, size_t ws_size,
                              hipStream_t stream) {
  const float* ctx   = (const float*)d_in[0];  // [B,H,C]
  const float* qn    = (const float*)d_in[1];  // [B,H,Q]
  const int*   cmask = (const int*)d_in[2];    // [B,C]
  const int*   qmask = (const int*)d_in[3];    // [B,Q]
  const float* w_c   = (const float*)d_in[4];  // [H]
  const float* w_q   = (const float*)d_in[5];  // [H]
  const float* w_cq  = (const float*)d_in[6];  // [H]
  const float* bias  = (const float*)d_in[7];  // scalar

  const size_t N1 = (size_t)BB * HH * CC;      // 8388608
  float* out0 = (float*)d_out;
  float* out1 = out0 + N1;
  float* out2 = out0 + 2 * N1;
  float* out3 = out0 + 3 * N1;

  // workspace layout (bytes)
  char* ws = (char*)d_ws;
  float*    S      = (float*)ws;                               // B*C*Q fp32   = 67108864 B
  _Float16* Srow   = (_Float16*)(ws + 67108864);               // B*C*Q f16    = 33554432 B
  _Float16* ScolT  = (_Float16*)(ws + 67108864 + 33554432);    // B*Q*C f16    = 33554432 B
  float*    tmp    = (float*)(ws + 67108864 + 2 * 33554432);   // B*Q*H fp32   =  4194304 B
  float*    part_c = (float*)(ws + 67108864 + 2 * 33554432 + 4194304);            // B*C fp32
  float*    part_q = (float*)(ws + 67108864 + 2 * 33554432 + 4194304 + 262144);   // B*Q fp32

  // K1: projections
  k_parts<<<(BB * CC + BB * QQ + 255) / 256, 256, 0, stream>>>(ctx, qn, w_c, w_q, part_c, part_q);

  // K2: similarity matrix S (TDM-staged, WMMA, K=H)
  // blocks = B * (C/64) * (Q/64) = 32*32*4 = 4096
  k_S<<<4096, 256, 0, stream>>>(ctx, qn, w_cq, part_c, part_q, bias, S);

  // K3a: row softmax over Q
  k_row_softmax<<<BB * CC, 256, 0, stream>>>(S, qmask, Srow);

  // K3b: column softmax over C (transposed output)
  k_col_softmax<<<BB * QQ, 256, 0, stream>>>(S, cmask, ScolT);

  // K4: c2q (WMMA, K=Q) + out0/out2
  k_c2q<<<8192, 128, 0, stream>>>(Srow, qn, ctx, out0, out1, out2);

  // K5: tmp = ScolT @ ctx^T (WMMA, K=C)
  k_tmp<<<1024, 128, 0, stream>>>(ScolT, ctx, tmp);

  // K6: q2c (WMMA, K=Q) -> out3
  k_q2c<<<8192, 128, 0, stream>>>(Srow, tmp, ctx, out3);
}
